// BuySellLinkPrediction_48842368090320
// MI455X (gfx1250) — matrix operand order repaired and checked by reading, not compiled, not run on hardware
//
#include <hip/hip_runtime.h>

#define NCN 5000
#define NTN 10000
#define NE  50000
#define NEL 20000
#define LAY 2

// msg_kernel staging geometry: 130 K-steps total = 26 phases x 5 K-steps
#define MSG_WPB    4      // waves per block
#define MSG_THREADS (MSG_WPB * 32)
#define MSG_CHUNK  5      // K-steps staged per phase (5 * 4KB = 20KB per buffer)
#define MSG_PHASES 26

typedef _Float16 h16;
typedef __attribute__((ext_vector_type(16))) _Float16 v16h;
typedef __attribute__((ext_vector_type(8)))  _Float16 v8h;
typedef __attribute__((ext_vector_type(8)))  float    v8f;

__device__ __forceinline__ v8f wmma32(v16h a, v16h b, v8f c) {
  return __builtin_amdgcn_wmma_f32_16x16x32_f16(false, a, false, b, (short)0, c, false, false);
}

// async DMA: global -> LDS, 16 bytes per lane, tracked by ASYNCcnt
__device__ __forceinline__ void async_load_b128(const void* gaddr, unsigned lds_off) {
  asm volatile("global_load_async_to_lds_b128 %0, %1, off"
               :: "v"(lds_off), "v"(gaddr)
               : "memory");
}
__device__ __forceinline__ void wait_asynccnt0() {
  asm volatile("s_wait_asynccnt 0x0" ::: "memory");
}

// float atomic-max via ordered-int trick; agg buffer is initialized to -inf.
__device__ __forceinline__ void atomicMaxF(float* addr, float v) {
  if (v >= 0.0f) atomicMax((int*)addr, __float_as_int(v));
  else           atomicMin((unsigned int*)addr, __float_as_uint(v));
}

// ---------------- weight swizzle: [K x 64] f32 row-major -> WMMA B fragments f16
// fragment f=(kt*4+nt): 512 halfs; per lane 16 contiguous halfs; element e:
// K = kt*32 + (lane>>4)*8 + (e&7) + (e>>3)*16, N = nt*16 + (lane&15)
__global__ void swizzle_w_kernel(const float* __restrict__ W, h16* __restrict__ out,
                                 int realK, int padK) {
  int t = blockIdx.x * blockDim.x + threadIdx.x;
  int total = padK * 64;
  if (t >= total) return;
  int e  = t & 15;
  int ln = (t >> 4) & 31;
  int f  = t >> 9;
  int kt = f >> 2, nt = f & 3;
  int K = kt * 32 + (ln >> 4) * 8 + (e & 7) + (e >> 3) * 16;
  int n = nt * 16 + (ln & 15);
  out[t] = (K < realK) ? (h16)W[K * 64 + n] : (h16)0.0f;
}

// conv mlp2 weight remap: kappa<4096: kh=kappa>>6, i=kappa&63 -> m2w[kh,i*64+o]
//                         kappa>=4096: bias rows -> m2b[(kappa-4096)*64+o]
__global__ void swizzle_m2_kernel(const float* __restrict__ m2w, const float* __restrict__ m2b,
                                  h16* __restrict__ out) {
  int t = blockIdx.x * blockDim.x + threadIdx.x;
  const int total = 4160 * 64;
  if (t >= total) return;
  int e  = t & 15;
  int ln = (t >> 4) & 31;
  int f  = t >> 9;
  int kt = f >> 2, nt = f & 3;
  int K = kt * 32 + (ln >> 4) * 8 + (e & 7) + (e >> 3) * 16;
  int n = nt * 16 + (ln & 15);
  float v;
  if (K < 4096) v = m2w[(size_t)(K >> 6) * 4096 + (size_t)(K & 63) * 64 + n];
  else          v = m2b[(size_t)(K - 4096) * 64 + n];
  out[t] = (h16)v;
}

__global__ void fill_neginf_kernel(float* __restrict__ p, int n) {
  int t = blockIdx.x * blockDim.x + threadIdx.x;
  if (t < n) p[t] = -__builtin_inff();
}

// ---------------- generic [M x 64] = act( A(gathered)[M x 64] @ Wfrag + bias + segmax(agg) )
__global__ void gemm64_kernel(const float* __restrict__ A, const int* __restrict__ gidx,
                              const h16* __restrict__ Wfrag, const float* __restrict__ bias,
                              const float* __restrict__ agg, float* __restrict__ out,
                              int M, int doRelu) {
  int g    = (blockIdx.x * blockDim.x + threadIdx.x) >> 5;
  int lane = threadIdx.x & 31;
  int ngroups = (M + 15) >> 4;
  if (g >= ngroups) return;
  int r = lane & 15, hs = lane >> 4;
  int row = g * 16 + r;
  if (row >= M) row = M - 1;
  int arow = gidx ? gidx[row] : row;
  const float* ap = A + (size_t)arow * 64;
  v8f acc[4] = {};
#pragma unroll
  for (int ks = 0; ks < 2; ++ks) {
    int k0 = ks * 32;
    v16h a;
#pragma unroll
    for (int e = 0; e < 8; ++e) a[e] = (h16)ap[k0 + hs * 8 + e];
#pragma unroll
    for (int e = 0; e < 8; ++e) a[8 + e] = (h16)ap[k0 + 16 + hs * 8 + e];
#pragma unroll
    for (int nt = 0; nt < 4; ++nt) {
      v16h b = *(const v16h*)(Wfrag + ((size_t)(ks * 4 + nt) * 32 + lane) * 16);
      acc[nt] = wmma32(a, b, acc[nt]);
    }
  }
#pragma unroll
  for (int nt = 0; nt < 4; ++nt) {
    int o = nt * 16 + (lane & 15);
    float bv = bias[o];
#pragma unroll
    for (int rr = 0; rr < 8; ++rr) {
      int m = g * 16 + rr + hs * 8;
      if (m < M) {
        float v = acc[nt][rr] + bv;
        if (agg) {
          float av = agg[(size_t)m * 64 + o];
          if (__float_as_uint(av) == 0xFF800000u) av = 0.0f;  // isolated node -> 0
          v += av;
        }
        if (doRelu) v = fmaxf(v, 0.0f);
        out[(size_t)m * 64 + o] = v;
      }
    }
  }
}

// ---------------- h1 = relu(eattr[E x 16] @ m1w + m1b) stored f16; K padded 16->32
__global__ void edge_mlp1_kernel(const float* __restrict__ eattr, const h16* __restrict__ W1frag,
                                 const float* __restrict__ b1, h16* __restrict__ h1g, int E) {
  int g    = (blockIdx.x * blockDim.x + threadIdx.x) >> 5;
  int lane = threadIdx.x & 31;
  int ngroups = E >> 4;
  if (g >= ngroups) return;
  int r = lane & 15, hs = lane >> 4;
  int e0 = g * 16;
  const float* ap = eattr + (size_t)(e0 + r) * 16;
  v16h a;
#pragma unroll
  for (int e = 0; e < 8; ++e) a[e] = (h16)ap[hs * 8 + e];   // K = hs*8+e < 16
#pragma unroll
  for (int e = 0; e < 8; ++e) a[8 + e] = (h16)0.0f;         // K >= 16 zero-padded
  v8f acc[4] = {};
#pragma unroll
  for (int nt = 0; nt < 4; ++nt) {
    v16h b = *(const v16h*)(W1frag + ((size_t)nt * 32 + lane) * 16);
    acc[nt] = wmma32(a, b, acc[nt]);
  }
#pragma unroll
  for (int nt = 0; nt < 4; ++nt) {
    int o = nt * 16 + (lane & 15);
    float bv = b1[o];
#pragma unroll
    for (int rr = 0; rr < 8; ++rr) {
      int e = e0 + rr + hs * 8;
      h1g[(size_t)e * 64 + o] = (h16)fmaxf(acc[nt][rr] + bv, 0.0f);
    }
  }
}

// ---------------- fused per-edge bilinear GEMM + segment-max scatter
// msg[16 x 64] = P[16 x 4160] @ W2frag[4160 x 64], P built on the fly:
// P[e, kappa] = h1[e, kappa>>6] * hsrc[e, kappa&63], with h1[e,64] = 1 for bias rows.
// B fragments are async-DMA'd into double-buffered LDS once per block and shared
// by MSG_WPB waves (cuts L2 traffic by MSG_WPB x vs per-wave streaming).
__global__ void __launch_bounds__(MSG_THREADS)
msg_kernel(const h16* __restrict__ h1g, const float* __restrict__ hsrcf,
           const int* __restrict__ esrc, const int* __restrict__ edst,
           const h16* __restrict__ W2frag, float* __restrict__ agg, int ngroups) {
  __shared__ __attribute__((aligned(16))) h16 shB[2][MSG_CHUNK * 2048];     // 2 x 20KB
  __shared__ __attribute__((aligned(16))) h16 sh_src[MSG_WPB][16 * 64];
  __shared__ h16 sh_h1[MSG_WPB][16 * 65];
  __shared__ int sh_dst[MSG_WPB][16];

  int tid  = threadIdx.x;
  int w    = tid >> 5;
  int lane = tid & 31;
  int gi   = blockIdx.x * MSG_WPB + w;
  int valid = gi < ngroups;
  int g = valid ? gi : (ngroups - 1);
  int e0 = g * 16;

  const char* gB = (const char*)W2frag;
  // kick off phase-0 B stage immediately
  {
    unsigned lbase = (unsigned)(size_t)(void*)&shB[0][0];
#pragma unroll
    for (int j = 0; j < (MSG_CHUNK * 4096) / (MSG_THREADS * 16); ++j) {
      unsigned off = (unsigned)(tid + j * MSG_THREADS) * 16u;
      async_load_b128(gB + off, lbase + off);
    }
  }
  // stage per-wave edge operands while the DMA runs
  for (int idx = lane; idx < 16 * 64; idx += 32) {
    int el = idx >> 6, i = idx & 63;
    sh_src[w][idx] = (h16)hsrcf[(size_t)esrc[e0 + el] * 64 + i];
    sh_h1[w][el * 65 + i] = h1g[(size_t)(e0 + el) * 64 + i];
  }
  if (lane < 16) {
    sh_h1[w][lane * 65 + 64] = (h16)1.0f;  // bias rows: h1 extended with 1
    sh_dst[w][lane] = edst[e0 + lane];
  }
  wait_asynccnt0();
  __syncthreads();

  int r = lane & 15, hs = lane >> 4;
  const h16* srow = &sh_src[w][r * 64];
  const h16* qrow = &sh_h1[w][r * 65];
  v8f acc[4] = {};
  for (int p = 0; p < MSG_PHASES; ++p) {
    int buf = p & 1;
    if (p + 1 < MSG_PHASES) {  // prefetch next phase into the idle buffer
      unsigned lbase = (unsigned)(size_t)(void*)&shB[buf ^ 1][0];
      const char* gsrc = gB + (size_t)(p + 1) * (MSG_CHUNK * 4096);
#pragma unroll
      for (int j = 0; j < (MSG_CHUNK * 4096) / (MSG_THREADS * 16); ++j) {
        unsigned off = (unsigned)(tid + j * MSG_THREADS) * 16u;
        async_load_b128(gsrc + off, lbase + off);
      }
    }
    const h16* bbuf = &shB[buf][0] + lane * 16;
#pragma unroll
    for (int kl = 0; kl < MSG_CHUNK; ++kl) {
      int kA = (p * MSG_CHUNK + kl) * 32 + hs * 8;  // 8-aligned, within one 64-block
      int kB = kA + 16;
      v8h s0 = *(const v8h*)(srow + (kA & 63));
      v8h s1 = *(const v8h*)(srow + (kB & 63));
      h16 q0 = qrow[kA >> 6];
      h16 q1 = qrow[kB >> 6];
      v8h a0 = s0 * q0;            // v_pk_mul_f16 x4
      v8h a1 = s1 * q1;
      v16h a = __builtin_shufflevector(a0, a1, 0, 1, 2, 3, 4, 5, 6, 7,
                                       8, 9, 10, 11, 12, 13, 14, 15);
#pragma unroll
      for (int nt = 0; nt < 4; ++nt) {
        v16h b = *(const v16h*)(bbuf + kl * 2048 + nt * 512);
        acc[nt] = wmma32(a, b, acc[nt]);
      }
    }
    wait_asynccnt0();
    __syncthreads();
  }
  if (valid) {
#pragma unroll
    for (int nt = 0; nt < 4; ++nt) {
      int o = nt * 16 + (lane & 15);
#pragma unroll
      for (int rr = 0; rr < 8; ++rr) {
        int d = sh_dst[w][rr + hs * 8];
        atomicMaxF(&agg[(size_t)d * 64 + o], acc[nt][rr]);
      }
    }
  }
}

// ---------------- prediction head: z=[hc|ht|eattr] (144->160) -> 64 -> 64 -> 1
__global__ void __launch_bounds__(32)
head_kernel(const float* __restrict__ hc, const float* __restrict__ ht,
            const int* __restrict__ lsrc, const int* __restrict__ ldst,
            const float* __restrict__ lattr,
            const h16* __restrict__ W0frag, const float* __restrict__ b0,
            const h16* __restrict__ W1frag, const float* __restrict__ b1,
            const float* __restrict__ w2, const float* __restrict__ b2,
            float* __restrict__ preds, float* __restrict__ logits) {
  __shared__ __attribute__((aligned(16))) h16 zsh[16 * 160];
  __shared__ __attribute__((aligned(16))) h16 hsh[16 * 64];
  int g = blockIdx.x, lane = threadIdx.x;
  int e0 = g * 16;
  for (int idx = lane; idx < 16 * 160; idx += 32) {
    int el = idx / 160, j = idx - el * 160;
    float v;
    if (j < 64)       v = hc[(size_t)lsrc[e0 + el] * 64 + j];
    else if (j < 128) v = ht[(size_t)ldst[e0 + el] * 64 + (j - 64)];
    else if (j < 144) v = lattr[(size_t)(e0 + el) * 16 + (j - 128)];
    else              v = 0.0f;
    zsh[idx] = (h16)v;
  }
  __syncthreads();
  int r = lane & 15, hs = lane >> 4, n = lane & 15;
  const h16* zrow = zsh + r * 160;
  v8f acc[4] = {};
#pragma unroll
  for (int ks = 0; ks < 5; ++ks) {
    int kA = ks * 32 + hs * 8;
    v8h a0 = *(const v8h*)(zrow + kA);
    v8h a1 = *(const v8h*)(zrow + kA + 16);
    v16h a = __builtin_shufflevector(a0, a1, 0, 1, 2, 3, 4, 5, 6, 7,
                                     8, 9, 10, 11, 12, 13, 14, 15);
#pragma unroll
    for (int nt = 0; nt < 4; ++nt) {
      v16h b = *(const v16h*)(W0frag + ((size_t)(ks * 4 + nt) * 32 + lane) * 16);
      acc[nt] = wmma32(a, b, acc[nt]);
    }
  }
#pragma unroll
  for (int nt = 0; nt < 4; ++nt) {
    int o = nt * 16 + n;
    float bv = b0[o];
#pragma unroll
    for (int rr = 0; rr < 8; ++rr)
      hsh[(size_t)(rr + hs * 8) * 64 + o] = (h16)fmaxf(acc[nt][rr] + bv, 0.0f);
  }
  __syncthreads();
  const h16* hrow = hsh + r * 64;
  v8f acc2[4] = {};
#pragma unroll
  for (int ks = 0; ks < 2; ++ks) {
    int kA = ks * 32 + hs * 8;
    v8h a0 = *(const v8h*)(hrow + kA);
    v8h a1 = *(const v8h*)(hrow + kA + 16);
    v16h a = __builtin_shufflevector(a0, a1, 0, 1, 2, 3, 4, 5, 6, 7,
                                     8, 9, 10, 11, 12, 13, 14, 15);
#pragma unroll
    for (int nt = 0; nt < 4; ++nt) {
      v16h b = *(const v16h*)(W1frag + ((size_t)(ks * 4 + nt) * 32 + lane) * 16);
      acc2[nt] = wmma32(a, b, acc2[nt]);
    }
  }
  float h2v[4][8];
  float w2c[4];
#pragma unroll
  for (int nt = 0; nt < 4; ++nt) {
    float bv = b1[nt * 16 + n];
    w2c[nt] = w2[nt * 16 + n];
#pragma unroll
    for (int rr = 0; rr < 8; ++rr) h2v[nt][rr] = fmaxf(acc2[nt][rr] + bv, 0.0f);
  }
  float bb = b2[0];
#pragma unroll
  for (int rr = 0; rr < 8; ++rr) {
    float p = 0.0f;
#pragma unroll
    for (int nt = 0; nt < 4; ++nt) p += h2v[nt][rr] * w2c[nt];
    p += __shfl_xor(p, 1);
    p += __shfl_xor(p, 2);
    p += __shfl_xor(p, 4);
    p += __shfl_xor(p, 8);
    if (n == 0) {
      int e = e0 + rr + hs * 8;
      float lg = p + bb;
      logits[e] = lg;
      preds[e] = 1.0f / (1.0f + __expf(-lg));
    }
  }
}

extern "C" void kernel_launch(void* const* d_in, const int* in_sizes, int n_in,
                              void* d_out, int out_size, void* d_ws, size_t ws_size,
                              hipStream_t stream) {
  (void)in_sizes; (void)n_in; (void)out_size; (void)ws_size;
  const int*   x_c    = (const int*)d_in[0];
  const int*   x_t    = (const int*)d_in[1];
  const int*   es_ct  = (const int*)d_in[2];
  const int*   ed_ct  = (const int*)d_in[3];
  const int*   es_tc  = (const int*)d_in[4];
  const int*   ed_tc  = (const int*)d_in[5];
  const int*   el_src = (const int*)d_in[6];
  const int*   el_dst = (const int*)d_in[7];
  const float* ea_ct  = (const float*)d_in[8];
  const float* ea_tc  = (const float*)d_in[9];
  const float* elattr = (const float*)d_in[10];
  const float* emb_c  = (const float*)d_in[11];
  const float* emb_t  = (const float*)d_in[12];
  const float* ntlcw  = (const float*)d_in[13];
  const float* ntlcb  = (const float*)d_in[14];
  const float* ntltw  = (const float*)d_in[15];
  const float* ntltb  = (const float*)d_in[16];
  const float* m1w    = (const float*)d_in[17];
  const float* m1b    = (const float*)d_in[18];
  const float* m2w    = (const float*)d_in[19];
  const float* m2b    = (const float*)d_in[20];
  const float* rw     = (const float*)d_in[21];
  const float* rb     = (const float*)d_in[22];
  const float* w0     = (const float*)d_in[23];
  const float* b0     = (const float*)d_in[24];
  const float* w1     = (const float*)d_in[25];
  const float* b1     = (const float*)d_in[26];
  const float* w2     = (const float*)d_in[27];
  const float* b2     = (const float*)d_in[28];

  char* wp = (char*)d_ws;
  auto alloc = [&](size_t b) { char* r = wp; wp += (b + 255) & ~(size_t)255; return (void*)r; };
  float* hc[2];  float* ht[2];
  hc[0] = (float*)alloc((size_t)NCN * 64 * 4);
  hc[1] = (float*)alloc((size_t)NCN * 64 * 4);
  ht[0] = (float*)alloc((size_t)NTN * 64 * 4);
  ht[1] = (float*)alloc((size_t)NTN * 64 * 4);
  float* agg_c = (float*)alloc((size_t)NCN * 64 * 4);
  float* agg_t = (float*)alloc((size_t)NTN * 64 * 4);
  h16* h1g    = (h16*)alloc((size_t)NE * 64 * 2);
  h16* ntlc_f = (h16*)alloc(64 * 64 * 2);
  h16* ntlt_f = (h16*)alloc(64 * 64 * 2);
  h16* m1f[4]; h16* m2f[4]; h16* rwf[4];
  for (int i = 0; i < 4; ++i) {
    m1f[i] = (h16*)alloc(32 * 64 * 2);
    m2f[i] = (h16*)alloc((size_t)4160 * 64 * 2);
    rwf[i] = (h16*)alloc(64 * 64 * 2);
  }
  h16* w0f = (h16*)alloc(160 * 64 * 2);
  h16* w1f = (h16*)alloc(64 * 64 * 2);

  auto swz = [&](const float* W, h16* o, int realK, int padK) {
    int total = padK * 64;
    swizzle_w_kernel<<<(total + 255) / 256, 256, 0, stream>>>(W, o, realK, padK);
  };
  swz(ntlcw, ntlc_f, 64, 64);
  swz(ntltw, ntlt_f, 64, 64);
  for (int i = 0; i < 4; ++i) {
    swz(m1w + (size_t)i * 16 * 64, m1f[i], 16, 32);
    swz(rw + (size_t)i * 64 * 64, rwf[i], 64, 64);
    swizzle_m2_kernel<<<(4160 * 64 + 255) / 256, 256, 0, stream>>>(
        m2w + (size_t)i * 64 * 4096, m2b + (size_t)i * 4096, m2f[i]);
  }
  swz(w0, w0f, 144, 160);
  swz(w1, w1f, 64, 64);

  auto gemm64 = [&](const float* A, const int* gidx, const h16* Wf, const float* bias,
                    const float* agg, float* o, int M, int doRelu) {
    int groups = (M + 15) / 16;
    gemm64_kernel<<<(groups + 7) / 8, 256, 0, stream>>>(A, gidx, Wf, bias, agg, o, M, doRelu);
  };

  const int ngroups = NE / 16;
  const int msg_blocks = (ngroups + MSG_WPB - 1) / MSG_WPB;

  // node-type linear transforms
  gemm64(emb_c, x_c, ntlc_f, ntlcb, nullptr, hc[0], NCN, 0);
  gemm64(emb_t, x_t, ntlt_f, ntltb, nullptr, ht[0], NTN, 0);

  int cur = 0;
  for (int l = 0; l < LAY; ++l) {
    int i0 = l * 2 + 0, i1 = l * 2 + 1;
    // c->t conv (produces new_t into agg_t)
    fill_neginf_kernel<<<(NTN * 64 + 255) / 256, 256, 0, stream>>>(agg_t, NTN * 64);
    edge_mlp1_kernel<<<(NE / 16 + 7) / 8, 256, 0, stream>>>(
        ea_ct, m1f[i0], m1b + (size_t)i0 * 64, h1g, NE);
    msg_kernel<<<msg_blocks, MSG_THREADS, 0, stream>>>(
        h1g, hc[cur], es_ct, ed_ct, m2f[i0], agg_t, ngroups);
    // t->c conv (uses pre-layer h_t)
    fill_neginf_kernel<<<(NCN * 64 + 255) / 256, 256, 0, stream>>>(agg_c, NCN * 64);
    edge_mlp1_kernel<<<(NE / 16 + 7) / 8, 256, 0, stream>>>(
        ea_tc, m1f[i1], m1b + (size_t)i1 * 64, h1g, NE);
    msg_kernel<<<msg_blocks, MSG_THREADS, 0, stream>>>(
        h1g, ht[cur], es_tc, ed_tc, m2f[i1], agg_c, ngroups);
    // finalize: relu(agg + h_dst @ root_w + root_b)
    gemm64(ht[cur], nullptr, rwf[i0], rb + (size_t)i0 * 64, agg_t, ht[1 - cur], NTN, 1);
    gemm64(hc[cur], nullptr, rwf[i1], rb + (size_t)i1 * 64, agg_c, hc[1 - cur], NCN, 1);
    cur = 1 - cur;
  }

  float* preds  = (float*)d_out;
  float* logitz = preds + NEL;
  head_kernel<<<NEL / 16, 32, 0, stream>>>(hc[cur], ht[cur], el_src, el_dst, elattr,
                                           w0f, b0, w1f, b1, w2, b2, preds, logitz);
}
